// VectorQuantizedVAE_25099788878022
// MI455X (gfx1250) — matrix operand order repaired and checked by reading, not compiled, not run on hardware
//
#include <hip/hip_runtime.h>
#include <cstddef>

typedef __attribute__((ext_vector_type(16))) _Float16 v16h;
typedef __attribute__((ext_vector_type(8)))  _Float16 v8h;
typedef __attribute__((ext_vector_type(8)))  float    v8f;
typedef __attribute__((ext_vector_type(4)))  unsigned int u32x4;
typedef __attribute__((ext_vector_type(8)))  int      i32x8;
typedef __attribute__((ext_vector_type(4)))  int      i32x4;

#if defined(__gfx1250__) && __has_builtin(__builtin_amdgcn_tensor_load_to_lds) && __has_builtin(__builtin_amdgcn_s_wait_tensorcnt)
#define HAVE_TDM 1
#else
#define HAVE_TDM 0
#endif
#if defined(__gfx1250__) && __has_builtin(__builtin_amdgcn_s_wait_asynccnt)
#define HAVE_ASYNC 1
#else
#define HAVE_ASYNC 0
#endif

constexpr int B_   = 16;
constexpr int T_   = 2400;
constexpr int TPH  = 600;
constexpr int H_   = 256;
constexpr int H2_  = 512;
constexpr int CQ_  = 128;
constexpr int MCB  = 512;   // codebook entries
constexpr int LDW  = 2448;  // padded time width: [2 zeros][2400 data][pad]
constexpr int NENC = 2432;  // GEMM N for encoder (19 x 128 tiles)
constexpr int GLD  = 640;   // padded phoneme width (5 x 128 tiles)

// ---------------- workspace layout (bytes) ----------------
constexpr size_t AL(size_t x){ return (x + 255) & ~size_t(255); }
constexpr size_t OFF_H     = 0;
constexpr size_t SZ_H      = (size_t)B_*H_*LDW*4;
constexpr size_t OFF_HLN   = AL(OFF_H + SZ_H);
constexpr size_t SZ_HLN    = (size_t)B_*H_*LDW*2;
constexpr size_t OFF_G     = AL(OFF_HLN + SZ_HLN);
constexpr size_t SZ_G      = (size_t)B_*H2_*LDW*2;
constexpr size_t OFF_NPAD  = AL(OFF_G + SZ_G);
constexpr size_t SZ_NP     = (size_t)B_*LDW*4;
constexpr size_t OFF_NPIN  = AL(OFF_NPAD + SZ_NP);
constexpr size_t OFF_GRP   = AL(OFF_NPIN + SZ_NP);
constexpr size_t SZ_GRP    = (size_t)B_*H_*GLD*4;
constexpr size_t OFF_CNT   = AL(OFF_GRP + SZ_GRP);
constexpr size_t SZ_CNT    = (size_t)B_*GLD*4;
constexpr size_t OFF_GRP16 = AL(OFF_CNT + SZ_CNT);
constexpr size_t SZ_GRP16  = (size_t)B_*H_*GLD*2;
constexpr size_t OFF_Z     = AL(OFF_GRP16 + SZ_GRP16);
constexpr size_t SZ_Z      = (size_t)B_*CQ_*GLD*4;
constexpr size_t OFF_Q16   = AL(OFF_Z + SZ_Z);
constexpr size_t SZ_Q16    = (size_t)B_*CQ_*GLD*2;
constexpr size_t OFF_LOSS  = AL(OFF_Q16 + SZ_Q16);
constexpr size_t ZERO_BYTES= AL(OFF_LOSS + 256);           // zero everything below here each call
constexpr size_t OFF_CBN   = ZERO_BYTES;
constexpr size_t OFF_W1    = AL(OFF_CBN + (size_t)MCB*4);  // [10][3][512][256] f16
constexpr size_t SZ_W1     = (size_t)10*3*H2_*H_*2;
constexpr size_t OFF_W2    = AL(OFF_W1 + SZ_W1);           // [10][256][512] f16
constexpr size_t SZ_W2     = (size_t)10*H_*H2_*2;
constexpr size_t OFF_WPOST = AL(OFF_W2 + SZ_W2);           // [3][256][256] f16
constexpr size_t SZ_WPOST  = (size_t)3*H_*H_*2;
constexpr size_t OFF_WPI   = AL(OFF_WPOST + SZ_WPOST);     // [128][256] f16
constexpr size_t OFF_WPO   = AL(OFF_WPI + (size_t)CQ_*H_*2); // [256][128] f16

constexpr size_t OUT_OFF_LOSS = (size_t)B_*H_*TPH;         // 2457600
constexpr size_t OUT_OFF_IDX  = OUT_OFF_LOSS + 1;

// async 16-byte global->LDS copy (ASYNCcnt-tracked). VDST = LDS byte address,
// VADDR = 64-bit global address, no SADDR.
__device__ __forceinline__ void async_copy_b128(const _Float16* gsrc, _Float16* ldst){
#if HAVE_ASYNC
  unsigned lds = (unsigned)(size_t)(__attribute__((address_space(3))) _Float16*)ldst;
  asm volatile("global_load_async_to_lds_b128 %0, %1, off"
               :: "v"(lds), "v"(gsrc) : "memory");
#else
  (void)gsrc; (void)ldst;
#endif
}

#if HAVE_TDM
// TDM: DMA a 128-row x 64-byte tile (row stride lda*2 bytes) into LDS with
// 16B padding after each 64B row (-> 80B LDS row stride). data_size=8B,
// tile_dim0=8 elem, tile_dim1=128, pad_interval=16 DWORDs, pad_amount=4 DWORDs.
__device__ __forceinline__ void tdm_load_A(const _Float16* gtile, _Float16* ldsDst, int lda){
  unsigned long long ga = (unsigned long long)(size_t)gtile;
  unsigned ldsb = (unsigned)(size_t)(__attribute__((address_space(3))) _Float16*)ldsDst;
  u32x4 g0;
  g0[0] = 1u;                                                  // count=1 (valid user D#)
  g0[1] = ldsb;                                                // lds_addr
  g0[2] = (unsigned)ga;                                        // global_addr[31:0]
  g0[3] = (unsigned)((ga >> 32) & 0x01FFFFFFu) | 0x80000000u;  // global_addr[56:32] | type=2
  i32x8 g1;
  g1[0] = (3<<16) | (1<<20) | (3<<22) | (3<<25);  // data_size=8B, pad_en, ivl=16dw, amt=4dw
  g1[1] = 8 << 16;                                 // tensor_dim0 = 8 (elements/row)
  g1[2] = 128 << 16;                               // tensor_dim1 = 128 (rows)
  g1[3] = 8 << 16;                                 // tile_dim0 = 8
  g1[4] = 128;                                     // tile_dim1 = 128, tile_dim2 = 0
  g1[5] = lda >> 2;                                // tensor_dim0_stride = lda*2B / 8B
  g1[6] = 0;
  g1[7] = 0;
  i32x4 gz = {0, 0, 0, 0};
#if __clang_major__ >= 23
  i32x8 gz8 = {0, 0, 0, 0, 0, 0, 0, 0};
  __builtin_amdgcn_tensor_load_to_lds(g0, g1, gz, gz, gz8, 0);
#else
  __builtin_amdgcn_tensor_load_to_lds(g0, g1, gz, gz, 0);
#endif
}
#endif

// ---------------- utility kernels ----------------
__global__ void zero_kernel(float4* p, size_t n4){
  size_t i = (size_t)blockIdx.x*blockDim.x + threadIdx.x;
  size_t st = (size_t)gridDim.x*blockDim.x;
  float4 z; z.x=0.f; z.y=0.f; z.z=0.f; z.w=0.f;
  for (; i < n4; i += st) p[i] = z;
}

__global__ void npin_kernel(const float* __restrict__ np, float* __restrict__ npin){
  int i = blockIdx.x*256 + threadIdx.x;          // i over B*T
  if (i >= B_*T_) return;
  int b = i / T_, t = i % T_;
  npin[(size_t)b*LDW + 2 + t] = np[i];
}

// k=3 conv weights [G][O][I][3] f32 -> [G][3][O][I] f16
__global__ void convert_conv_kernel(const float* __restrict__ in, _Float16* __restrict__ out,
                                    int OC, int IC, int n){
  int i = blockIdx.x*256 + threadIdx.x;
  if (i >= n) return;
  int c  = i % IC;  int t1 = i / IC;
  int o  = t1 % OC; int t2 = t1 / OC;
  int tp = t2 % 3;  int g  = t2 / 3;
  out[i] = (_Float16) in[ (((size_t)g*OC + o)*IC + c)*3 + tp ];
}

__global__ void convert_plain_kernel(const float* __restrict__ in, _Float16* __restrict__ out, int n){
  int i = blockIdx.x*256 + threadIdx.x;
  if (i < n) out[i] = (_Float16)in[i];
}

__global__ void cbnorm_kernel(const float* __restrict__ cb, float* __restrict__ cbn){
  int m = blockIdx.x*256 + threadIdx.x;
  if (m >= MCB) return;
  float s = 0.f;
  const float* r = cb + (size_t)m*CQ_;
  for (int k = 0; k < CQ_; ++k) s += r[k]*r[k];
  cbn[m] = s;
}

// 1x1 conv_in: x[:, :20, :] -> h[B,256,*], masked by in_nonpadding
__global__ __launch_bounds__(256) void conv_in_kernel(const float* __restrict__ x,
    const float* __restrict__ w, const float* __restrict__ bias,
    const float* __restrict__ np, float* __restrict__ h){
  int o = threadIdx.x, t = blockIdx.x, b = blockIdx.y;
  float acc = bias[o];
  const float* xc = x + (size_t)b*80*T_ + t;
  const float* wr = w + o*20;
  #pragma unroll
  for (int i = 0; i < 20; ++i) acc += wr[i] * xc[(size_t)i*T_];
  acc *= np[(size_t)b*T_ + t];
  h[((size_t)b*H_ + o)*LDW + 2 + t] = acc;
}

// channel LayerNorm (dim=1): one wave per timestep; optional npad computation + optional pre/post mask
__global__ __launch_bounds__(256) void ln_kernel(const float* __restrict__ h,
    _Float16* __restrict__ out, const float* __restrict__ gamma, const float* __restrict__ beta,
    float eps, float* __restrict__ npadOut, const float* __restrict__ maskPad){
  int wave = threadIdx.x >> 5, lane = threadIdx.x & 31;
  int t = blockIdx.x*8 + wave;   // grid.x = 300 -> t in [0,2400)
  int b = blockIdx.y;
  const float* col = h + (size_t)b*H_*LDW + 2 + t;
  float mk = maskPad ? maskPad[(size_t)b*LDW + 2 + t] : 1.f;
  float vals[8];
  float s = 0.f, sq = 0.f, sa = 0.f;
  #pragma unroll
  for (int i = 0; i < 8; ++i){
    float v = col[(size_t)(lane + 32*i)*LDW] * mk;
    vals[i] = v; s += v; sq += v*v; sa += fabsf(v);
  }
  #pragma unroll
  for (int off = 16; off > 0; off >>= 1){
    s  += __shfl_xor(s,  off, 32);
    sq += __shfl_xor(sq, off, 32);
    sa += __shfl_xor(sa, off, 32);
  }
  float mean = s * (1.f/H_);
  float varv = sq * (1.f/H_) - mean*mean;
  float rstd = rsqrtf(varv + eps);
  _Float16* ocol = out + (size_t)b*H_*LDW + 2 + t;
  #pragma unroll
  for (int i = 0; i < 8; ++i){
    int c = lane + 32*i;
    float v = (vals[i] - mean)*rstd*gamma[c] + beta[c];
    if (maskPad) v *= mk;
    ocol[(size_t)c*LDW] = (_Float16)v;
  }
  if (npadOut && lane == 0)
    npadOut[(size_t)b*LDW + 2 + t] = (sa > 0.f) ? 1.f : 0.f;
}

// ---------------- generic f16 WMMA GEMM, double-buffered, fused epilogues ----------------
// C[M,N] = sum_tap A_tap[M,Kd] * B[Kd, n+tap], f32 accumulate.
// Block 256 thr = 8 waves; tile BM=128 BN=128 BK=32; wave grid 4x2; wave tile 32x64 = 2x4 wmma.
// A tile: TDM descriptor DMA (TENSORcnt) if available, else async-to-LDS (ASYNCcnt), else sync.
// B tile: 8kx2n register transpose -> two ds_store_b128 per thread (tap shift is 2B-aligned only).
__global__ __launch_bounds__(256) void gemm_f16_wmma(
    const _Float16* __restrict__ A, int lda, int aTapStride,
    const _Float16* __restrict__ B0, int ldb, size_t bStride,
    float* __restrict__ C32, _Float16* __restrict__ C16, int ldc, size_t cStride, int cColOff,
    const float* __restrict__ bias,
    const float* __restrict__ resid, size_t rStride,
    const float* __restrict__ npad, int npadLd,
    int Kd, int taps, float scale, int act, int nmax)
{
  __shared__ __align__(16) _Float16 As[2][128*40];   // [m][k], row stride 40 halves (80B)
  __shared__ __align__(16) _Float16 Bs[2][128*40];   // [n][k] transposed, row stride 40
  const int tid = threadIdx.x;
  const int n0 = blockIdx.x*128, m0 = blockIdx.y*128, b = blockIdx.z;
  const _Float16* Bb = B0 + (size_t)b*bStride;
  const int wv = tid >> 5, lane = tid & 31;
  const int wm = wv >> 1, wn = wv & 1;
  const int lr = lane & 15, kh = lane >> 4;
  const int ksteps = Kd >> 5;
  const int S = taps * ksteps;

  const int bko = (tid >> 6) * 8;      // B staging: k rows bko..bko+7
  const int bnc = (tid & 63) * 2;      // B staging: 2 n values

  auto aTile = [&](int s)->const _Float16*{
    int tap = s / ksteps, k0 = (s - tap*ksteps) << 5;
    return A + (size_t)tap*aTapStride + (size_t)m0*lda + k0;
  };
  auto bSrc = [&](int s)->const _Float16*{
    int tap = s / ksteps, k0 = (s - tap*ksteps) << 5;
    return Bb + (size_t)(k0 + bko)*ldb + n0 + bnc + tap;
  };

  v8f acc[2][4] = {};

  // ---- prologue: stage tile 0 into buffer 0 ----
  {
#if HAVE_TDM
    if (wv == 0) tdm_load_A(aTile(0), &As[0][0], lda);
#elif HAVE_ASYNC
    #pragma unroll
    for (int part = 0; part < 2; ++part){
      int id = tid + part*256;
      int r = id >> 2, ch = (id & 3)*8;
      async_copy_b128(aTile(0) + (size_t)r*lda + ch, &As[0][r*40 + ch]);
    }
#else
    #pragma unroll
    for (int part = 0; part < 2; ++part){
      int id = tid + part*256;
      int r = id >> 2, ch = (id & 3)*8;
      *(v8h*)&As[0][r*40 + ch] = *(const v8h*)(aTile(0) + (size_t)r*lda + ch);
    }
#endif
    const _Float16* bs = bSrc(0);
    _Float16 bq[2][8];
    #pragma unroll
    for (int r = 0; r < 8; ++r){
      bq[0][r] = bs[(size_t)r*ldb];
      bq[1][r] = bs[(size_t)r*ldb + 1];
    }
    #pragma unroll
    for (int i = 0; i < 2; ++i){
      v8h v;
      #pragma unroll
      for (int r = 0; r < 8; ++r) v[r] = bq[i][r];
      *(v8h*)&Bs[0][(bnc + i)*40 + bko] = v;
    }
  }

  // ---- main pipelined loop: one barrier per K-step ----
  for (int s = 0; s < S; ++s){
    int cur = s & 1, nxt = cur ^ 1;
    bool hasNext = (s + 1 < S);

    // prefetch next tile's B (8k x 2n micro-tile) into registers before the barrier
    _Float16 bq[2][8];
#if !HAVE_TDM && !HAVE_ASYNC
    v8h aR0, aR1;
#endif
    if (hasNext){
#if !HAVE_TDM && !HAVE_ASYNC
      {
        int id = tid;
        int r = id >> 2, ch = (id & 3)*8;
        aR0 = *(const v8h*)(aTile(s + 1) + (size_t)r*lda + ch);
        id = tid + 256; r = id >> 2; ch = (id & 3)*8;
        aR1 = *(const v8h*)(aTile(s + 1) + (size_t)r*lda + ch);
      }
#endif
      const _Float16* bs = bSrc(s + 1);
      #pragma unroll
      for (int r = 0; r < 8; ++r){
        bq[0][r] = bs[(size_t)r*ldb];
        bq[1][r] = bs[(size_t)r*ldb + 1];
      }
    }
    if (s + 2 < S) __builtin_prefetch(bSrc(s + 2), 0, 0);

#if HAVE_TDM
    if (wv == 0) __builtin_amdgcn_s_wait_tensorcnt(0);   // A tile for buf 'cur' landed
#elif HAVE_ASYNC
    __builtin_amdgcn_s_wait_asynccnt(0);
#endif
    __syncthreads();                        // buf 'cur' staged; buf 'nxt' free

#if HAVE_TDM
    if (hasNext && wv == 0) tdm_load_A(aTile(s + 1), &As[nxt][0], lda);
#elif HAVE_ASYNC
    if (hasNext){
      #pragma unroll
      for (int part = 0; part < 2; ++part){
        int id = tid + part*256;
        int r = id >> 2, ch = (id & 3)*8;
        async_copy_b128(aTile(s + 1) + (size_t)r*lda + ch, &As[nxt][r*40 + ch]);
      }
    }
#endif

    // fragments per documented 16-bit A 16x32 / B 32x16 wave32 layouts
    v16h af[2], bf[4];
    #pragma unroll
    for (int i = 0; i < 2; ++i){
      const v8h* arow = (const v8h*)&As[cur][(wm*32 + i*16 + lr)*40];
      ((v8h*)&af[i])[0] = arow[kh];        // K = kh*8 .. +7
      ((v8h*)&af[i])[1] = arow[2 + kh];    // K = 16+kh*8 .. +7
    }
    #pragma unroll
    for (int j = 0; j < 4; ++j){
      const v8h* brow = (const v8h*)&Bs[cur][(wn*64 + j*16 + lr)*40];
      ((v8h*)&bf[j])[0] = brow[kh];        // K = kh*8 .. +7
      ((v8h*)&bf[j])[1] = brow[2 + kh];    // K = 16+kh*8 .. +7
    }
    #pragma unroll
    for (int i = 0; i < 2; ++i)
      #pragma unroll
      for (int j = 0; j < 4; ++j)
        acc[i][j] = __builtin_amdgcn_wmma_f32_16x16x32_f16(
            false, af[i], false, bf[j], (short)0, acc[i][j], false, false);

    // commit staged registers into buf 'nxt'
    if (hasNext){
#if !HAVE_TDM && !HAVE_ASYNC
      {
        int id = tid;
        int r = id >> 2, ch = (id & 3)*8;
        *(v8h*)&As[nxt][r*40 + ch] = aR0;
        id = tid + 256; r = id >> 2; ch = (id & 3)*8;
        *(v8h*)&As[nxt][r*40 + ch] = aR1;
      }
#endif
      #pragma unroll
      for (int i = 0; i < 2; ++i){
        v8h v;
        #pragma unroll
        for (int r = 0; r < 8; ++r) v[r] = bq[i][r];
        *(v8h*)&Bs[nxt][(bnc + i)*40 + bko] = v;
      }
    }
  }

  // ---- epilogue: lane -> (n = lr), VGPR r -> (m = kh*8 + r) ----
  #pragma unroll
  for (int i = 0; i < 2; ++i)
    #pragma unroll
    for (int j = 0; j < 4; ++j){
      int nn = n0 + wn*64 + j*16 + lr;
      if (nn >= nmax) continue;
      float np = npad ? npad[(size_t)b*npadLd + cColOff + nn] : 1.f;
      #pragma unroll
      for (int r = 0; r < 8; ++r){
        int mm = m0 + wm*32 + i*16 + kh*8 + r;
        float v = (acc[i][j][r] + bias[mm]) * scale;
        if (act) v = 0.5f*v*(1.f + erff(v*0.70710678118654752f));   // exact GELU
        size_t ci = (size_t)b*cStride + (size_t)mm*ldc + cColOff + nn;
        if (resid) v += resid[(size_t)b*rStride + (size_t)mm*ldc + cColOff + nn];
        v *= np;
        if (C32) C32[ci] = v;
        if (C16) C16[ci] = (_Float16)v;
      }
    }
}

// ---------------- grouping, VQ, finalize ----------------
__global__ __launch_bounds__(256) void group_kernel(const float* __restrict__ h,
    const int* __restrict__ mel2ph, float* __restrict__ grp, float* __restrict__ cnt){
  int t = blockIdx.x, b = blockIdx.y, c = threadIdx.x;
  int ph = mel2ph[(size_t)b*T_ + t];
  if (ph <= 0 || ph > TPH) return;          // segment 0 = padding, dropped
  float v = h[((size_t)b*H_ + c)*LDW + 2 + t];
  atomicAdd(&grp[((size_t)b*H_ + c)*GLD + (ph - 1)], v);
  if (c == 0) atomicAdd(&cnt[(size_t)b*GLD + (ph - 1)], 1.f);
}

__global__ void divide_kernel(const float* __restrict__ grp, const float* __restrict__ cnt,
                              _Float16* __restrict__ g16, float* __restrict__ g32){
  int i = blockIdx.x*256 + threadIdx.x;     // over B*H*GLD
  int p  = i % GLD;
  int bc = i / GLD;
  int b  = bc / H_;
  float v = grp[i] / fmaxf(cnt[(size_t)b*GLD + p], 1.f);
  g32[i] = v;
  g16[i] = (_Float16)v;
}

__global__ __launch_bounds__(256) void vq_kernel(const float* __restrict__ z,
    const float* __restrict__ cb, const float* __restrict__ cbn,
    _Float16* __restrict__ q16, float* __restrict__ lossAcc, float* __restrict__ idxOut){
  __shared__ float zcol[8][CQ_];
  int wave = threadIdx.x >> 5, lane = threadIdx.x & 31;
  int tp = blockIdx.x*8 + wave;             // grid.x = 75 -> tp in [0,600)
  int b  = blockIdx.y;
  const float* zc = z + (size_t)b*CQ_*GLD + tp;
  float zn = 0.f;
  #pragma unroll
  for (int i = 0; i < 4; ++i){
    float v = zc[(size_t)(lane + 32*i)*GLD];
    zcol[wave][lane + 32*i] = v;
    zn += v*v;
  }
  __syncthreads();
  float bestd = 3.4e38f; int bestm = 0;
  for (int j = 0; j < MCB/32; ++j){
    int m = lane + 32*j;
    const float* cw = cb + (size_t)m*CQ_;
    float dot = 0.f;
    for (int k = 0; k < CQ_; ++k) dot += zcol[wave][k]*cw[k];
    float d = cbn[m] - 2.f*dot;             // ||z-c||^2 - ||z||^2
    if (d < bestd || (d == bestd && m < bestm)){ bestd = d; bestm = m; }
  }
  #pragma unroll
  for (int off = 16; off > 0; off >>= 1){
    float od = __shfl_xor(bestd, off, 32);
    int   om = __shfl_xor(bestm, off, 32);
    if (od < bestd || (od == bestd && om < bestm)){ bestd = od; bestm = om; }
    zn += __shfl_xor(zn, off, 32);
  }
  // gather straight-through q = codebook[idx]
  const float* cw = cb + (size_t)bestm*CQ_;
  _Float16* qc = q16 + (size_t)b*CQ_*GLD + tp;
  #pragma unroll
  for (int i = 0; i < 4; ++i){
    int c = lane + 32*i;
    qc[(size_t)c*GLD] = (_Float16)cw[c];
  }
  if (lane == 0){
    atomicAdd(lossAcc, zn + bestd);         // = ||z - q||^2 for this (b,tp)
    idxOut[(size_t)b*TPH + tp] = (float)bestm;
  }
}

__global__ void finalize_loss_kernel(const float* acc, float* out){
  out[0] = acc[0] * (0.25f / ((float)B_ * (float)TPH * (float)CQ_));
}

// ---------------- host-side launch ----------------
extern "C" void kernel_launch(void* const* d_in, const int* in_sizes, int n_in,
                              void* d_out, int out_size, void* d_ws, size_t ws_size,
                              hipStream_t stream) {
  (void)in_sizes; (void)n_in; (void)out_size; (void)ws_size;
  const float* x       = (const float*)d_in[0];
  const float* np_in   = (const float*)d_in[2];
  const int*   mel2ph  = (const int*)  d_in[4];
  const float* ciw     = (const float*)d_in[7];
  const float* cib     = (const float*)d_in[8];
  const float* ln_g    = (const float*)d_in[9];
  const float* ln_b    = (const float*)d_in[10];
  const float* w1      = (const float*)d_in[11];
  const float* b1      = (const float*)d_in[12];
  const float* w2      = (const float*)d_in[13];
  const float* b2      = (const float*)d_in[14];
  const float* lastg   = (const float*)d_in[15];
  const float* lastb   = (const float*)d_in[16];
  const float* postw   = (const float*)d_in[17];
  const float* postb   = (const float*)d_in[18];
  const float* piw     = (const float*)d_in[19];
  const float* pib     = (const float*)d_in[20];
  const float* cb      = (const float*)d_in[21];
  const float* pow_    = (const float*)d_in[22];
  const float* pob     = (const float*)d_in[23];

  char* ws = (char*)d_ws;
  float*     Hbuf  = (float*)    (ws + OFF_H);
  _Float16*  HLN   = (_Float16*) (ws + OFF_HLN);
  _Float16*  Gbuf  = (_Float16*) (ws + OFF_G);
  float*     NPAD  = (float*)    (ws + OFF_NPAD);
  float*     NPIN  = (float*)    (ws + OFF_NPIN);
  float*     GRP   = (float*)    (ws + OFF_GRP);
  float*     CNT   = (float*)    (ws + OFF_CNT);
  _Float16*  GRP16 = (_Float16*) (ws + OFF_GRP16);
  float*     Zbuf  = (float*)    (ws + OFF_Z);
  _Float16*  Q16   = (_Float16*) (ws + OFF_Q16);
  float*     LOSS  = (float*)    (ws + OFF_LOSS);
  float*     CBN   = (float*)    (ws + OFF_CBN);
  _Float16*  W1F   = (_Float16*) (ws + OFF_W1);
  _Float16*  W2F   = (_Float16*) (ws + OFF_W2);
  _Float16*  WPF   = (_Float16*) (ws + OFF_WPOST);
  _Float16*  WPIF  = (_Float16*) (ws + OFF_WPI);
  _Float16*  WPOF  = (_Float16*) (ws + OFF_WPO);
  float*     OUT   = (float*)d_out;

  // 1) zero all activation scratch (pad columns must be zero; atomics need zero init)
  zero_kernel<<<2048, 256, 0, stream>>>((float4*)ws, ZERO_BYTES/16);
  // 2) padded input nonpadding
  npin_kernel<<<(B_*T_ + 255)/256, 256, 0, stream>>>(np_in, NPIN);
  // 3) weight conversions to f16
  convert_conv_kernel<<<(10*3*H2_*H_)/256, 256, 0, stream>>>(w1, W1F, H2_, H_, 10*3*H2_*H_);
  convert_conv_kernel<<<(3*H_*H_)/256,    256, 0, stream>>>(postw, WPF, H_, H_, 3*H_*H_);
  convert_plain_kernel<<<(10*H_*H2_)/256, 256, 0, stream>>>(w2, W2F, 10*H_*H2_);
  convert_plain_kernel<<<(CQ_*H_)/256,    256, 0, stream>>>(piw, WPIF, CQ_*H_);
  convert_plain_kernel<<<(H_*CQ_)/256,    256, 0, stream>>>(pow_, WPOF, H_*CQ_);
  cbnorm_kernel<<<2, 256, 0, stream>>>(cb, CBN);
  // 4) conv_in 20->256 (tiny K, VALU) -> h
  conv_in_kernel<<<dim3(T_, B_), 256, 0, stream>>>(x, ciw, cib, np_in, Hbuf);

  const float invsqrt3 = 0.57735026918962576f;
  // 5) 5 residual blocks x 2 sub-layers
  for (int blk = 0; blk < 5; ++blk){
    for (int j = 0; j < 2; ++j){
      int bj = blk*2 + j;
      ln_kernel<<<dim3(300, B_), 256, 0, stream>>>(Hbuf, HLN, ln_g + bj*H_, ln_b + bj*H_,
                                                   1e-5f, (j == 0) ? NPAD : nullptr, nullptr);
      // conv1: 512 x 2432 x (256 x 3 taps), bias, *3^-1/2, GELU -> g (f16)
      gemm_f16_wmma<<<dim3(NENC/128, H2_/128, B_), 256, 0, stream>>>(
          W1F + (size_t)bj*3*H2_*H_, H_, H2_*H_,
          HLN, LDW, (size_t)H_*LDW,
          nullptr, Gbuf, LDW, (size_t)H2_*LDW, 2,
          b1 + bj*H2_, nullptr, 0, nullptr, 0,
          H_, 3, invsqrt3, 1, NENC);
      // conv2 1x1: 256 x 2432 x 512, +bias, +residual(h), *npad -> h (f32)
      gemm_f16_wmma<<<dim3(NENC/128, H_/128, B_), 256, 0, stream>>>(
          W2F + (size_t)bj*H_*H2_, H2_, 0,
          Gbuf + 2, LDW, (size_t)H2_*LDW,
          Hbuf, nullptr, LDW, (size_t)H_*LDW, 2,
          b2 + bj*H_, Hbuf, (size_t)H_*LDW, NPAD, LDW,
          H2_, 1, 1.f, 0, NENC);
    }
  }
  // 6) final LN (eps 1e-12) with in_nonpadding pre+post mask
  ln_kernel<<<dim3(300, B_), 256, 0, stream>>>(Hbuf, HLN, lastg, lastb, 1e-12f, nullptr, NPIN);
  // 7) post conv k=3: 256 x 2432 x (256 x 3), *in_nonpadding -> h (= ph_z_e)
  gemm_f16_wmma<<<dim3(NENC/128, H_/128, B_), 256, 0, stream>>>(
      WPF, H_, H_*H_,
      HLN, LDW, (size_t)H_*LDW,
      Hbuf, nullptr, LDW, (size_t)H_*LDW, 2,
      postb, nullptr, 0, NPIN, LDW,
      H_, 3, 1.f, 0, NENC);
  // 8) group frames by phoneme segment (atomic mean)
  group_kernel<<<dim3(T_, B_), 256, 0, stream>>>(Hbuf, mel2ph, GRP, CNT);
  divide_kernel<<<(B_*H_*GLD)/256, 256, 0, stream>>>(GRP, CNT, GRP16, GRP);
  // 9) proj_in: 128 x 640 x 256 -> z
  gemm_f16_wmma<<<dim3(GLD/128, CQ_/128, B_), 256, 0, stream>>>(
      WPIF, H_, 0,
      GRP16, GLD, (size_t)H_*GLD,
      Zbuf, nullptr, GLD, (size_t)CQ_*GLD, 0,
      pib, nullptr, 0, nullptr, 0,
      H_, 1, 1.f, 0, GLD);
  // 10) VQ: argmin over 512 codewords, gather q, loss, indices
  vq_kernel<<<dim3(TPH/8, B_), 256, 0, stream>>>(Zbuf, cb, CBN, Q16, LOSS, OUT + OUT_OFF_IDX);
  // 11) proj_out: 256 x 600 x 128 -> d_out
  gemm_f16_wmma<<<dim3(GLD/128, H_/128, B_), 256, 0, stream>>>(
      WPOF, CQ_, 0,
      Q16, GLD, (size_t)CQ_*GLD,
      OUT, nullptr, TPH, (size_t)H_*TPH, 0,
      pob, nullptr, 0, nullptr, 0,
      CQ_, 1, 1.f, 0, TPH);
  // 12) loss scalar
  finalize_loss_kernel<<<1, 1, 0, stream>>>(LOSS, OUT + OUT_OFF_LOSS);
}
